// SelfAttention_56770877718999
// MI455X (gfx1250) — matrix operand order repaired and checked by reading, not compiled, not run on hardware
//
#include <hip/hip_runtime.h>

// Problem constants (reference: B=4, S=4096, D=768, fp32)
#define BB 4
#define SS 4096
#define DD 768
#define NROWS (BB * SS) // 16384

typedef __attribute__((ext_vector_type(16))) __bf16 v16bf;
typedef __attribute__((ext_vector_type(8)))  float  v8f;

union FragB16 { v16bf v; uint4 u[2]; };

__device__ __forceinline__ __bf16 f2bf(float f) {
    // native f32 -> bf16 conversion (backend selects v_cvt_*bf16* ops)
    return (__bf16)f;
}

__device__ __forceinline__ float fast_exp2(float x) {
    return __builtin_exp2f(x);   // lowers to v_exp_f32
}

__device__ __forceinline__ v8f wmma_bf16(v16bf a, v16bf b, v8f c) {
    return __builtin_amdgcn_wmma_f32_16x16x32_bf16(
        /*neg_a=*/false, a, /*neg_b=*/false, b,
        /*c_mod=*/(short)0, c, /*reuse_a=*/false, /*reuse_b=*/false);
}

// ---------------------------------------------------------------------------
// Phase 1: Q = x Wq^T, K = x Wk^T, V = x Wv^T  (fp32 in -> bf16 out)
//   Q, K stored row-major [NROWS, DD] bf16.
//   V stored TRANSPOSED per batch: Vt[b][e][s]  ([DD, SS] bf16 per batch),
//   so phase-2 B-fragments are contiguous loads.
// One wave computes a 16x128 output strip (8 column tiles): the A fragment
// (x rows, fp32->bf16) is loaded/converted once per k-chunk and reused for
// 8 WMMAs. 24 k-chunks * 8 tiles = 192 bf16 WMMAs per wave.
// ---------------------------------------------------------------------------
__global__ __launch_bounds__(256) void qkv_proj_kernel(
    const float* __restrict__ x,
    const float* __restrict__ Wq,
    const float* __restrict__ Wk,
    const float* __restrict__ Wv,
    __bf16* __restrict__ Qb,
    __bf16* __restrict__ Kb,
    __bf16* __restrict__ Vt)
{
    const int lane = threadIdx.x & 31;
    const int wave = threadIdx.x >> 5;
    const int half = lane >> 4;   // 0 or 1
    const int lm   = lane & 15;

    // tasks: 3 matrices * 1024 row-blocks * 6 column-groups = 18432; 8/block
    const int TPM = (NROWS / 16) * (DD / 128);   // 6144 per matrix
    int task = blockIdx.x * 8 + wave;
    int mat  = task / TPM;                       // 0=Q 1=K 2=V
    int rem  = task % TPM;
    int tm   = rem / (DD / 128);
    int cg   = rem % (DD / 128);
    const float* W = (mat == 0) ? Wq : ((mat == 1) ? Wk : Wv);
    const int i0 = tm * 16;       // output row base (flat batch*seq)
    const int eb = cg * 128;      // output column-group base

    v8f acc[8];
    #pragma unroll
    for (int t = 0; t < 8; t++) { v8f z = {}; acc[t] = z; }

    for (int kb = 0; kb < DD; kb += 32) {
        // A fragment: x rows (fp32 -> bf16). Lane lm holds row m = lm.
        // Runs: [kb + c .. +7], [kb + c + 16 .. +23],  c = half ? 8 : 0
        FragB16 a;
        {
            const float* xp = x + (long)(i0 + lm) * DD + kb + (half ? 8 : 0);
            float4 p0 = *(const float4*)(xp);
            float4 p1 = *(const float4*)(xp + 4);
            float4 p2 = *(const float4*)(xp + 16);
            float4 p3 = *(const float4*)(xp + 20);
            float fa[16] = { p0.x,p0.y,p0.z,p0.w, p1.x,p1.y,p1.z,p1.w,
                             p2.x,p2.y,p2.z,p2.w, p3.x,p3.y,p3.z,p3.w };
            #pragma unroll
            for (int q = 0; q < 16; q++) a.v[q] = f2bf(fa[q]);
        }
        // 8 B fragments, one per 16-wide column tile; reuse A for each.
        #pragma unroll
        for (int ct = 0; ct < 8; ct++) {
            const int e0 = eb + 16 * ct;
            // B(k=d, n=e) = W[e][d]. Lane holds col n = e0+lm,
            // runs: [kb + c .. +7], [kb + c + 16 .. +23], c = half ? 8 : 0. (fp32 -> bf16)
            FragB16 bfr;
            const float* wp = W + (long)(e0 + lm) * DD + kb + (half ? 8 : 0);
            float4 p0 = *(const float4*)(wp);
            float4 p1 = *(const float4*)(wp + 4);
            float4 p2 = *(const float4*)(wp + 16);
            float4 p3 = *(const float4*)(wp + 20);
            float fb[16] = { p0.x,p0.y,p0.z,p0.w, p1.x,p1.y,p1.z,p1.w,
                             p2.x,p2.y,p2.z,p2.w, p3.x,p3.y,p3.z,p3.w };
            #pragma unroll
            for (int q = 0; q < 16; q++) bfr.v[q] = f2bf(fb[q]);
            acc[ct] = wmma_bf16(a.v, bfr.v, acc[ct]);
        }
    }

    // C layout: row m = v + 8*half, col n = lm
    if (mat < 2) {
        __bf16* outp = (mat == 0) ? Qb : Kb;
        #pragma unroll
        for (int ct = 0; ct < 8; ct++) {
            const int e0 = eb + 16 * ct;
            #pragma unroll
            for (int v = 0; v < 8; v++) {
                int m = v + 8 * half;
                outp[(long)(i0 + m) * DD + e0 + lm] = f2bf(acc[ct][v]);
            }
        }
    } else {
        int b  = i0 / SS;   // 16-row tile never crosses a batch (4096 % 16 == 0)
        int s0 = i0 % SS;
        #pragma unroll
        for (int ct = 0; ct < 8; ct++) {
            const int e0 = eb + 16 * ct;
            #pragma unroll
            for (int v = 0; v < 8; v++) {
                int m = v + 8 * half;
                Vt[(long)b * DD * SS + (long)(e0 + lm) * SS + (s0 + m)] = f2bf(acc[ct][v]);
            }
        }
    }
}

// ---------------------------------------------------------------------------
// Phase 2: flash attention, causal, online softmax.
// One block (8 waves) per 16-query tile. Wave w owns feature slice
// [96w, 96w+96): computes partial scores over its slice, deterministic LDS
// reduction, softmax stats by 16 threads, every wave rebuilds P and does PV
// for its 6 output column tiles.
// ---------------------------------------------------------------------------
__global__ __launch_bounds__(256) void attn_kernel(
    const __bf16* __restrict__ Qb,
    const __bf16* __restrict__ Kb,
    const __bf16* __restrict__ Vt,
    float* __restrict__ out)
{
    __shared__ float sc_part[8][16][33];  // per-wave partial 16x32 scores (padded)
    __shared__ float sc[16][33];          // reduced scores
    __shared__ float st_m[16], st_l[16], st_alpha[16];

    const int lane = threadIdx.x & 31;
    const int wave = threadIdx.x >> 5;
    const int half = lane >> 4;
    const int lm   = lane & 15;

    const int b  = blockIdx.x >> 8;        // batch
    const int q0 = (blockIdx.x & 255) * 16;
    const long rowbase = (long)b * SS;     // flat row base into Qb/Kb
    const int f0 = wave * 96;              // this wave's feature slice

    // softmax scale folded with log2(e): exp(x*s) = exp2(x*s*log2e)
    const float SC = 0.036084391824351615f * 1.4426950408889634f;

    if (threadIdx.x < 16) { st_m[threadIdx.x] = -3.0e38f; st_l[threadIdx.x] = 0.0f; }

    // Preload Q A-fragments for this wave's slice (3 chunks of k=32)
    FragB16 qf[3];
    #pragma unroll
    for (int c = 0; c < 3; c++) {
        const __bf16* qp = Qb + (rowbase + q0 + lm) * DD + f0 + 32 * c + (half ? 8 : 0);
        qf[c].u[0] = *(const uint4*)(qp);
        qf[c].u[1] = *(const uint4*)(qp + 16);
    }

    v8f o[6];
    #pragma unroll
    for (int t = 0; t < 6; t++) { v8f z = {}; o[t] = z; }

    for (int kb = 0; kb <= q0 + 15; kb += 32) {
        // ---- partial QK^T over this wave's 96-feature slice ----
        v8f s0v = {}, s1v = {};
        #pragma unroll
        for (int c = 0; c < 3; c++) {
            // B(k=feat, n=key) = K[key][feat]; feat runs [c..c+7],[c+16..c+23], c = half ? 8 : 0
            FragB16 k0, k1;
            const __bf16* kp0 = Kb + (rowbase + kb + lm) * DD + f0 + 32 * c + (half ? 8 : 0);
            const __bf16* kp1 = Kb + (rowbase + kb + 16 + lm) * DD + f0 + 32 * c + (half ? 8 : 0);
            k0.u[0] = *(const uint4*)(kp0); k0.u[1] = *(const uint4*)(kp0 + 16);
            k1.u[0] = *(const uint4*)(kp1); k1.u[1] = *(const uint4*)(kp1 + 16);
            s0v = wmma_bf16(qf[c].v, k0.v, s0v);
            s1v = wmma_bf16(qf[c].v, k1.v, s1v);
        }
        #pragma unroll
        for (int v = 0; v < 8; v++) {
            sc_part[wave][v + 8 * half][lm]      = s0v[v];
            sc_part[wave][v + 8 * half][16 + lm] = s1v[v];
        }
        __syncthreads();

        // ---- deterministic cross-wave reduction (256 threads, 2 elems each) ----
        {
            int base = threadIdx.x * 2;
            #pragma unroll
            for (int rep = 0; rep < 2; rep++) {
                int e = base + rep;
                int r = e >> 5, cc = e & 31;
                float s = 0.0f;
                #pragma unroll
                for (int w = 0; w < 8; w++) s += sc_part[w][r][cc];
                sc[r][cc] = s;
            }
        }
        __syncthreads();

        // ---- online-softmax stats (threads 0..15, one per query row) ----
        if (threadIdx.x < 16) {
            int r = threadIdx.x;
            int qrow = q0 + r;
            float mo = st_m[r];
            float mx = mo;
            for (int j = 0; j < 32; j++)
                if (kb + j <= qrow) mx = fmaxf(mx, sc[r][j]);
            float alpha = fast_exp2((mo - mx) * SC);
            float sum = 0.0f;
            for (int j = 0; j < 32; j++)
                if (kb + j <= qrow) sum += fast_exp2((sc[r][j] - mx) * SC);
            st_m[r] = mx;
            st_alpha[r] = alpha;
            st_l[r] = st_l[r] * alpha + sum;
        }
        __syncthreads();

        // ---- build P in A-fragment layout (row m = lm; key runs per half) ----
        FragB16 pf;
        {
            int qrow = q0 + lm;
            float mnew = st_m[lm];
            int c0 = half ? 8 : 0;
            #pragma unroll
            for (int q = 0; q < 8; q++) {
                int j = c0 + q;
                float p = (kb + j <= qrow) ? fast_exp2((sc[lm][j] - mnew) * SC) : 0.0f;
                pf.v[q] = f2bf(p);
                int j2 = c0 + 16 + q;
                float p2 = (kb + j2 <= qrow) ? fast_exp2((sc[lm][j2] - mnew) * SC) : 0.0f;
                pf.v[8 + q] = f2bf(p2);
            }
        }

        // ---- rescale O and accumulate P x V for our 6 column tiles ----
        float av[8];
        #pragma unroll
        for (int v = 0; v < 8; v++) av[v] = st_alpha[v + 8 * half];
        #pragma unroll
        for (int t = 0; t < 6; t++) {
            // B(k=key, n=feat) = Vt[b][feat][key]; key runs [c..c+7],[c+16..c+23], c = half ? 8 : 0
            FragB16 vf;
            const __bf16* vp = Vt + (long)b * DD * SS
                             + (long)(f0 + 16 * t + lm) * SS + kb + (half ? 8 : 0);
            vf.u[0] = *(const uint4*)(vp); vf.u[1] = *(const uint4*)(vp + 16);
            v8f on;
            #pragma unroll
            for (int v = 0; v < 8; v++) on[v] = o[t][v] * av[v];
            o[t] = wmma_bf16(pf.v, vf.v, on);
        }
        // No extra barrier needed: next iteration's first __syncthreads()
        // orders sc_part rewrites after all reads of sc/st_*.
    }

    // ---- epilogue: normalize by row sums and store fp32 ----
    float lrec[8];
    #pragma unroll
    for (int v = 0; v < 8; v++) lrec[v] = 1.0f / st_l[v + 8 * half];
    #pragma unroll
    for (int t = 0; t < 6; t++) {
        #pragma unroll
        for (int v = 0; v < 8; v++) {
            int m = v + 8 * half;
            out[((long)(b * SS + q0 + m)) * DD + f0 + 16 * t + lm] = o[t][v] * lrec[v];
        }
    }
}

// ---------------------------------------------------------------------------
extern "C" void kernel_launch(void* const* d_in, const int* in_sizes, int n_in,
                              void* d_out, int out_size, void* d_ws, size_t ws_size,
                              hipStream_t stream) {
    (void)in_sizes; (void)n_in; (void)out_size; (void)ws_size;
    const float* x  = (const float*)d_in[0];
    const float* Wq = (const float*)d_in[1];
    const float* Wk = (const float*)d_in[2];
    const float* Wv = (const float*)d_in[3];
    float* out = (float*)d_out;

    // Workspace: Q [NROWS,DD] bf16, K [NROWS,DD] bf16, Vt [BB][DD][SS] bf16
    // = 3 * 16384 * 768 * 2 bytes = 75,497,472 bytes
    __bf16* Qb = (__bf16*)d_ws;
    __bf16* Kb = Qb + (size_t)NROWS * DD;
    __bf16* Vt = Kb + (size_t)NROWS * DD;

    // Phase 1: 3 * 1024 row-blocks * 6 column-groups = 18432 wave-tasks, 8/block
    const int tasks = 3 * (NROWS / 16) * (DD / 128);
    qkv_proj_kernel<<<tasks / 8, 256, 0, stream>>>(x, Wq, Wk, Wv, Qb, Kb, Vt);

    // Phase 2: one block per (batch, 16-query tile)
    attn_kernel<<<BB * (SS / 16), 256, 0, stream>>>(Qb, Kb, Vt, out);
}
